// Net_33492154974647
// MI455X (gfx1250) — compile-verified, matrix operand order checked by reading, not run on hardware
//
#include <hip/hip_runtime.h>
#include <math.h>

// ---------------------------------------------------------------------------
// PointTransformer forward for gfx1250 (MI455X).
// All K>=32 linear layers run through V_WMMA_F32_16X16X32_BF16.
// A and B are pre-converted to bf16 once per GEMM (B transposed) so the WMMA
// inner loop is pure b128 loads + wmma. KNN candidate tiles are staged to LDS
// with gfx1250 async-to-LDS ops when the builtins are available.
// Param pointers are consumed in setup_inputs() dict insertion order:
//   x, pos,
//   mlp_input{W,b,gamma,beta},
//   t_in{lin_in{W,b}, lin_out{W,b}, pos_nn[{W,b},{W,b}], attn_nn[{W,b},{W,b}],
//        lin.W, lin_src.W, lin_dst.W},
//   mlp_down[0..3]{W,b,gamma,beta},
//   t_down[0..3]{...}, head[0..2]{W,b}
// ---------------------------------------------------------------------------

typedef __attribute__((ext_vector_type(16))) __bf16 v16bf;
typedef __attribute__((ext_vector_type(8)))  __bf16 v8bf;
typedef __attribute__((ext_vector_type(8)))  float  v8f;

#define GF_BIAS 1
#define GF_RELU 2
#define GF_ADD  4

#define AS1 __attribute__((address_space(1)))
#define AS3 __attribute__((address_space(3)))

#if defined(__gfx1250__) && \
    __has_builtin(__builtin_amdgcn_global_load_async_to_lds_b32) && \
    __has_builtin(__builtin_amdgcn_s_wait_asynccnt)
#define HAVE_ASYNC_COPY 1
#else
#define HAVE_ASYNC_COPY 0
#endif

// ---------------- bf16 conversion helpers (hoisted out of the GEMM) ---------
__global__ void cvt_bf16_kernel(const float* __restrict__ in,
                                __bf16* __restrict__ out, int n) {
  int t = blockIdx.x * 256 + threadIdx.x;
  if (t < n) out[t] = (__bf16)in[t];
}

// W[K,Nc] (row-major) -> BT[Nc,K] bf16, so B fragments are contiguous loads.
__global__ void cvtT_bf16_kernel(const float* __restrict__ W,
                                 __bf16* __restrict__ BT, int K, int Nc) {
  int t = blockIdx.x * 256 + threadIdx.x;
  if (t >= K * Nc) return;
  int k = t / Nc, n = t % Nc;
  BT[(size_t)n * K + k] = (__bf16)W[t];
}

// ---------------- WMMA GEMM: C[M,Nc] = A[M,K] @ B[K,Nc] (+bias, relu, +skip) -
// One wave per 16x32 output tile (one A fragment feeds two WMMAs).
// A is bf16 [M,K]; BT is bf16 [Nc,K]. M %16==0, K %32==0, Nc %32==0.
__global__ void gemm_wmma_kernel(const __bf16* __restrict__ A,
                                 const __bf16* __restrict__ BT,
                                 const float* __restrict__ bias,
                                 const float* __restrict__ skip,
                                 float* __restrict__ C,
                                 int M, int Kd, int Nc, int flags) {
  const int lane = threadIdx.x & 31;
  const int h = lane >> 4;        // half-wave selects K-group
  const int r = lane & 15;        // row (A) / col (B) within tile
  const int arow = blockIdx.x * 16 + r;
  const int bcol = blockIdx.y * 32 + r;

  const __bf16* ap  = A  + (size_t)arow * Kd + h * 8;
  const __bf16* bp0 = BT + (size_t)bcol * Kd + h * 8;
  const __bf16* bp1 = bp0 + (size_t)16 * Kd;

  v8f acc0 = {}, acc1 = {};
  for (int k0 = 0; k0 < Kd; k0 += 32) {
    if (k0 + 32 < Kd) {
      __builtin_prefetch(ap + 32, 0, 1);    // global_prefetch_b8
      __builtin_prefetch(bp0 + 32, 0, 1);
      __builtin_prefetch(bp1 + 32, 0, 1);
    }
    v8bf alo = *(const v8bf*)ap;            // K = k0 + 8h + 0..7
    v8bf ahi = *(const v8bf*)(ap + 16);     // K = k0 + 16 + 8h + 0..7
    v8bf b0l = *(const v8bf*)bp0;
    v8bf b0h = *(const v8bf*)(bp0 + 16);
    v8bf b1l = *(const v8bf*)bp1;
    v8bf b1h = *(const v8bf*)(bp1 + 16);
    v16bf a  = __builtin_shufflevector(alo, ahi, 0,1,2,3,4,5,6,7,8,9,10,11,12,13,14,15);
    v16bf b0 = __builtin_shufflevector(b0l, b0h, 0,1,2,3,4,5,6,7,8,9,10,11,12,13,14,15);
    v16bf b1 = __builtin_shufflevector(b1l, b1h, 0,1,2,3,4,5,6,7,8,9,10,11,12,13,14,15);
    acc0 = __builtin_amdgcn_wmma_f32_16x16x32_bf16(false, a, false, b0,
                                                   (short)0, acc0, false, false);
    acc1 = __builtin_amdgcn_wmma_f32_16x16x32_bf16(false, a, false, b1,
                                                   (short)0, acc1, false, false);
    ap += 32; bp0 += 32; bp1 += 32;
  }

  const int oc0 = blockIdx.y * 32 + (lane & 15);
  const int oc1 = oc0 + 16;
  const float bv0 = (flags & GF_BIAS) ? bias[oc0] : 0.0f;
  const float bv1 = (flags & GF_BIAS) ? bias[oc1] : 0.0f;
#pragma unroll
  for (int e = 0; e < 8; ++e) {
    const int orow = blockIdx.x * 16 + e + 8 * h;  // C/D layout: vgpr e, half h
    float v0 = acc0[e] + bv0;
    float v1 = acc1[e] + bv1;
    if (flags & GF_RELU) { v0 = fmaxf(v0, 0.0f); v1 = fmaxf(v1, 0.0f); }
    if (flags & GF_ADD) {
      v0 += skip[(size_t)orow * Nc + oc0];
      v1 += skip[(size_t)orow * Nc + oc1];
    }
    C[(size_t)orow * Nc + oc0] = v0;
    C[(size_t)orow * Nc + oc1] = v1;
  }
}

// ---------------- K=3 linear (mlp_input first layer) -------------------------
__global__ void lin3_kernel(const float* __restrict__ X, const float* __restrict__ W,
                            const float* __restrict__ b, float* __restrict__ O,
                            int N, int C) {
  int t = blockIdx.x * 256 + threadIdx.x;
  if (t >= N * C) return;
  int i = t / C, c = t % C;
  O[t] = b[c] + X[i * 3] * W[c] + X[i * 3 + 1] * W[C + c] + X[i * 3 + 2] * W[2 * C + c];
}

// ---------------- BatchNorm (training stats, biased var) ---------------------
__global__ void bn_stats_kernel(const float* __restrict__ H, int N, int C,
                                float* __restrict__ mean, float* __restrict__ inv) {
  __shared__ float s1[256], s2[256];
  const int c = blockIdx.x, tid = threadIdx.x;
  float a = 0.f, b = 0.f;
  for (int i = tid; i < N; i += 256) {
    float v = H[(size_t)i * C + c];
    a += v; b += v * v;
  }
  s1[tid] = a; s2[tid] = b;
  __syncthreads();
  for (int st = 128; st > 0; st >>= 1) {
    if (tid < st) { s1[tid] += s1[tid + st]; s2[tid] += s2[tid + st]; }
    __syncthreads();
  }
  if (tid == 0) {
    float m = s1[0] / (float)N;
    float v = s2[0] / (float)N - m * m;
    mean[c] = m;
    inv[c] = rsqrtf(v + 1e-5f);
  }
}

__global__ void bn_relu_kernel(const float* __restrict__ H, const float* __restrict__ mean,
                               const float* __restrict__ inv, const float* __restrict__ gamma,
                               const float* __restrict__ beta, float* __restrict__ O,
                               int N, int C) {
  int t = blockIdx.x * 256 + threadIdx.x;
  if (t >= N * C) return;
  int c = t % C;
  float v = (H[t] - mean[c]) * inv[c] * gamma[c] + beta[c];
  O[t] = fmaxf(v, 0.0f);
}

// ---------------- KNN (LDS candidate tiling, top-16 per query) ---------------
#define KNN 16
__global__ void knn_kernel(int Q, const float* __restrict__ qpos,
                           int Nc, const float* __restrict__ cpos,
                           int* __restrict__ out, int ostride,
                           int selfcol, int excl) {
  __shared__ float tile[256 * 3];
  const int t = blockIdx.x * 256 + threadIdx.x;
  const bool valid = (t < Q);
  float qx = 0.f, qy = 0.f, qz = 0.f;
  if (valid) { qx = qpos[t * 3]; qy = qpos[t * 3 + 1]; qz = qpos[t * 3 + 2]; }

  float bd[KNN]; int bi[KNN];
#pragma unroll
  for (int k = 0; k < KNN; ++k) { bd[k] = 3.0e38f; bi[k] = 0; }
  float worst = 3.0e38f; int wslot = 0;

  for (int base = 0; base < Nc; base += 256) {
    __syncthreads();
    int ci = base + threadIdx.x;
#if HAVE_ASYNC_COPY
    if (ci < Nc) {
      AS3 int* dst = (AS3 int*)&tile[threadIdx.x * 3];
      AS1 int* src = (AS1 int*)(cpos + ci * 3);
      __builtin_amdgcn_global_load_async_to_lds_b32(src,     dst,     0, 0);
      __builtin_amdgcn_global_load_async_to_lds_b32(src + 1, dst + 1, 0, 0);
      __builtin_amdgcn_global_load_async_to_lds_b32(src + 2, dst + 2, 0, 0);
    }
    __builtin_amdgcn_s_wait_asynccnt(0);
#else
    if (ci < Nc) {
      tile[threadIdx.x * 3]     = cpos[ci * 3];
      tile[threadIdx.x * 3 + 1] = cpos[ci * 3 + 1];
      tile[threadIdx.x * 3 + 2] = cpos[ci * 3 + 2];
    }
#endif
    __syncthreads();
    if (valid) {
      int lim = Nc - base; if (lim > 256) lim = 256;
      for (int j = 0; j < lim; ++j) {
        int cj = base + j;
        if (excl && cj == t) continue;
        float dx = qx - tile[j * 3];
        float dy = qy - tile[j * 3 + 1];
        float dz = qz - tile[j * 3 + 2];
        float d2 = dx * dx + dy * dy + dz * dz;
        if (d2 < worst) {
          bd[wslot] = d2; bi[wslot] = cj;
          worst = bd[0]; wslot = 0;
#pragma unroll
          for (int k = 1; k < KNN; ++k)
            if (bd[k] > worst) { worst = bd[k]; wslot = k; }
        }
      }
    }
  }
  if (valid) {
    int* row = out + (size_t)t * ostride;
    if (selfcol) { row[0] = t; row += 1; }
#pragma unroll
    for (int k = 0; k < KNN; ++k) row[k] = bi[k];
  }
}

// ---------------- Farthest point sampling (single block, LDS dists) ----------
__global__ void fps_kernel(const float* __restrict__ pos, int N, int M,
                           int* __restrict__ sel) {
  __shared__ float dists[8192];
  __shared__ float rv[1024];
  __shared__ int ri[1024];
  __shared__ int s_last;
  const int tid = threadIdx.x;
  for (int i = tid; i < N; i += 1024) dists[i] = 3.0e38f;
  if (tid == 0) { sel[0] = 0; s_last = 0; }
  __syncthreads();

  for (int s = 1; s < M; ++s) {
    const int last = s_last;
    const float lx = pos[last * 3], ly = pos[last * 3 + 1], lz = pos[last * 3 + 2];
    float bv = -1.0f; int bidx = 0x7fffffff;
    for (int i = tid; i < N; i += 1024) {
      float dx = pos[i * 3] - lx, dy = pos[i * 3 + 1] - ly, dz = pos[i * 3 + 2] - lz;
      float d = dx * dx + dy * dy + dz * dz;
      float nd = fminf(dists[i], d);
      dists[i] = nd;
      if (nd > bv || (nd == bv && i < bidx)) { bv = nd; bidx = i; }
    }
    rv[tid] = bv; ri[tid] = bidx;
    __syncthreads();
    for (int st = 512; st > 0; st >>= 1) {
      if (tid < st) {
        if (rv[tid + st] > rv[tid] ||
            (rv[tid + st] == rv[tid] && ri[tid + st] < ri[tid])) {
          rv[tid] = rv[tid + st]; ri[tid] = ri[tid + st];
        }
      }
      __syncthreads();
    }
    if (tid == 0) { sel[s] = ri[0]; s_last = ri[0]; }
    __syncthreads();
  }
}

__global__ void gather_pos_kernel(const float* __restrict__ pos,
                                  const int* __restrict__ sel, int M,
                                  float* __restrict__ out) {
  int m = blockIdx.x * 256 + threadIdx.x;
  if (m >= M) return;
  int s = sel[m];
  out[m * 3] = pos[s * 3]; out[m * 3 + 1] = pos[s * 3 + 1]; out[m * 3 + 2] = pos[s * 3 + 2];
}

// ---------------- transition_down pooling: per-row max over K neighbors ------
__global__ void gather_max_kernel(const float* __restrict__ H,
                                  const int* __restrict__ nbr, int M, int C,
                                  float* __restrict__ O) {
  int t = blockIdx.x * 256 + threadIdx.x;
  if (t >= M * C) return;
  int m = t / C, c = t % C;
  float mx = -3.0e38f;
#pragma unroll
  for (int k = 0; k < KNN; ++k) {
    int j = nbr[m * KNN + k];
    mx = fmaxf(mx, H[(size_t)j * C + c]);
  }
  O[t] = mx;
}

// ---------------- per-edge pos_nn layer 1 (K=3 -> 64, relu) -----------------
__global__ void edge_posnn1_kernel(const float* __restrict__ pos,
                                   const int* __restrict__ idx17,
                                   int chunkStart, int cN,
                                   const float* __restrict__ W1,
                                   const float* __restrict__ b1,
                                   float* __restrict__ hid) {
  int t = blockIdx.x * 256 + threadIdx.x;
  int total = cN * 17 * 64;
  if (t >= total) return;
  int hc = t & 63;
  int e = t >> 6;
  int li = e / 17, k = e % 17;
  int i = chunkStart + li;
  int j = idx17[(size_t)i * 17 + k];
  float dx = pos[i * 3]     - pos[j * 3];
  float dy = pos[i * 3 + 1] - pos[j * 3 + 1];
  float dz = pos[i * 3 + 2] - pos[j * 3 + 2];
  float v = b1[hc] + dx * W1[hc] + dy * W1[64 + hc] + dz * W1[128 + hc];
  hid[t] = fmaxf(v, 0.0f);
}

// ---------------- attn input: a_dst_i - a_src_j + delta ----------------------
__global__ void edge_attn_in_kernel(const float* __restrict__ adst,
                                    const float* __restrict__ asrc,
                                    const float* __restrict__ delta,
                                    const int* __restrict__ idx17,
                                    int chunkStart, int cN, int C,
                                    float* __restrict__ out) {
  int t = blockIdx.x * 256 + threadIdx.x;
  int total = cN * 17 * C;
  if (t >= total) return;
  int c = t % C;
  int e = t / C;
  int li = e / 17, k = e % 17;
  int i = chunkStart + li;
  int j = idx17[(size_t)i * 17 + k];
  out[t] = adst[(size_t)i * C + c] - asrc[(size_t)j * C + c] + delta[t];
}

// ---------------- softmax over 17 edges + weighted aggregate -----------------
__global__ void edge_agg_kernel(const float* __restrict__ attn,
                                const float* __restrict__ delta,
                                const float* __restrict__ vv,
                                const int* __restrict__ idx17,
                                int chunkStart, int cN, int C,
                                float* __restrict__ out) {
  int t = blockIdx.x * 256 + threadIdx.x;
  if (t >= cN * C) return;
  int c = t % C, li = t / C;
  int i = chunkStart + li;
  float mx = -3.0e38f;
  for (int k = 0; k < 17; ++k)
    mx = fmaxf(mx, attn[((size_t)(li * 17 + k)) * C + c]);
  float s = 0.f, acc = 0.f;
  for (int k = 0; k < 17; ++k) {
    size_t e = (size_t)(li * 17 + k);
    float w = __expf(attn[e * C + c] - mx);
    int j = idx17[(size_t)i * 17 + k];
    acc += w * (vv[(size_t)j * C + c] + delta[e * C + c]);
    s += w;
  }
  out[(size_t)i * C + c] = acc / s;
}

// ---------------- global mean pool + 3-layer head + log_softmax --------------
__global__ void head_kernel(const float* __restrict__ X, int N,
                            const float* __restrict__ W0, const float* __restrict__ b0,
                            const float* __restrict__ W1, const float* __restrict__ b1,
                            const float* __restrict__ W2, const float* __restrict__ b2,
                            float* __restrict__ out) {
  __shared__ float feat[512], h0[64], h1[64], lg[40];
  const int t = threadIdx.x;  // block 512
  if (t < 512) {
    float s = 0.f;
    for (int i = 0; i < N; ++i) s += X[(size_t)i * 512 + t];
    feat[t] = s / (float)N;
  }
  __syncthreads();
  if (t < 64) {
    float s = b0[t];
    for (int c = 0; c < 512; ++c) s += feat[c] * W0[(size_t)c * 64 + t];
    h0[t] = fmaxf(s, 0.0f);
  }
  __syncthreads();
  if (t < 64) {
    float s = b1[t];
    for (int c = 0; c < 64; ++c) s += h0[c] * W1[c * 64 + t];
    h1[t] = fmaxf(s, 0.0f);
  }
  __syncthreads();
  if (t < 40) {
    float s = b2[t];
    for (int c = 0; c < 64; ++c) s += h1[c] * W2[c * 40 + t];
    lg[t] = s;
  }
  __syncthreads();
  if (t == 0) {
    float mx = lg[0];
    for (int k = 1; k < 40; ++k) mx = fmaxf(mx, lg[k]);
    float s = 0.f;
    for (int k = 0; k < 40; ++k) s += __expf(lg[k] - mx);
    float ls = __logf(s);
    for (int k = 0; k < 40; ++k) out[k] = lg[k] - mx - ls;
  }
}

// ===========================================================================
// Host orchestration
// ===========================================================================
struct Lin { const float* W; const float* b; };
struct TB {
  Lin lin_in, lin_out, pos0, pos1, attn0, attn1;
  const float *Wv, *Wsrc, *Wdst;
};
struct MB { Lin lin; const float *gamma, *beta; };

static Lin readLin(void* const* d_in, int& t, bool bias) {
  Lin l;
  l.W = (const float*)d_in[t++];
  l.b = bias ? (const float*)d_in[t++] : nullptr;
  return l;
}
static MB readMB(void* const* d_in, int& t) {
  MB m;
  m.lin = readLin(d_in, t, true);
  m.gamma = (const float*)d_in[t++];
  m.beta  = (const float*)d_in[t++];
  return m;
}
static TB readTB(void* const* d_in, int& t) {
  TB tb;
  tb.lin_in  = readLin(d_in, t, true);
  tb.lin_out = readLin(d_in, t, true);
  tb.pos0 = readLin(d_in, t, true);
  tb.pos1 = readLin(d_in, t, true);
  tb.attn0 = readLin(d_in, t, true);
  tb.attn1 = readLin(d_in, t, true);
  tb.Wv   = (const float*)d_in[t++];
  tb.Wsrc = (const float*)d_in[t++];
  tb.Wdst = (const float*)d_in[t++];
  return tb;
}

static inline unsigned gdiv(int n) { return (unsigned)((n + 255) / 256); }

struct Bufs {
  float *h, *adst, *asrc, *vv, *conv;
  float *eHid, *eDelta, *eAttn;
  __bf16 *Abf, *BTbf;
};

// cvtA==false reuses the previously converted A (same pointer/shape).
static void gemm(const float* A, const float* Bw, const float* bias,
                 const float* skip, float* Cout,
                 int M, int K, int Nc, int flags, bool cvtA,
                 const Bufs& B, hipStream_t st) {
  if (cvtA)
    cvt_bf16_kernel<<<gdiv(M * K), 256, 0, st>>>(A, B.Abf, M * K);
  cvtT_bf16_kernel<<<gdiv(K * Nc), 256, 0, st>>>(Bw, B.BTbf, K, Nc);
  dim3 g((unsigned)(M / 16), (unsigned)(Nc / 32));
  gemm_wmma_kernel<<<g, 32, 0, st>>>(B.Abf, B.BTbf, bias, skip, Cout, M, K, Nc, flags);
}

static void run_tb(const TB& tb, float* X, const float* pos, const int* nbr17,
                   int N, int C, const Bufs& B, hipStream_t st) {
  gemm(X,   tb.lin_in.W, tb.lin_in.b, nullptr, B.h,    N, C, C, GF_BIAS | GF_RELU, true,  B, st);
  gemm(B.h, tb.Wdst,     nullptr,     nullptr, B.adst, N, C, C, 0,                 true,  B, st);
  gemm(B.h, tb.Wsrc,     nullptr,     nullptr, B.asrc, N, C, C, 0,                 false, B, st);
  gemm(B.h, tb.Wv,       nullptr,     nullptr, B.vv,   N, C, C, 0,                 false, B, st);

  for (int cs = 0; cs < N; cs += 2048) {
    int cN = N - cs; if (cN > 2048) cN = 2048;
    const int E = cN * 17;
    edge_posnn1_kernel<<<gdiv(E * 64), 256, 0, st>>>(pos, nbr17, cs, cN,
                                                     tb.pos0.W, tb.pos0.b, B.eHid);
    gemm(B.eHid, tb.pos1.W, tb.pos1.b, nullptr, B.eDelta, E, 64, C, GF_BIAS, true, B, st);
    edge_attn_in_kernel<<<gdiv(E * C), 256, 0, st>>>(B.adst, B.asrc, B.eDelta,
                                                     nbr17, cs, cN, C, B.eAttn);
    gemm(B.eAttn, tb.attn0.W, tb.attn0.b, nullptr, B.eHid,  E, C, 64, GF_BIAS | GF_RELU, true, B, st);
    gemm(B.eHid,  tb.attn1.W, tb.attn1.b, nullptr, B.eAttn, E, 64, C, GF_BIAS,          true, B, st);
    edge_agg_kernel<<<gdiv(cN * C), 256, 0, st>>>(B.eAttn, B.eDelta, B.vv,
                                                  nbr17, cs, cN, C, B.conv);
  }
  // out = relu(lin_out(conv)) + skip, written in-place into X (same-thread RMW)
  gemm(B.conv, tb.lin_out.W, tb.lin_out.b, X, X, N, C, C,
       GF_BIAS | GF_RELU | GF_ADD, true, B, st);
}

extern "C" void kernel_launch(void* const* d_in, const int* in_sizes, int n_in,
                              void* d_out, int out_size, void* d_ws, size_t ws_size,
                              hipStream_t stream) {
  (void)in_sizes; (void)n_in; (void)out_size; (void)ws_size;

  // ---- parse params in setup_inputs() dict insertion order ----
  int t = 0;
  const float* X0  = (const float*)d_in[t++];
  const float* POS = (const float*)d_in[t++];
  MB mi = readMB(d_in, t);
  TB tin = readTB(d_in, t);
  MB md[4]; for (int i = 0; i < 4; ++i) md[i] = readMB(d_in, t);
  TB td[4]; for (int i = 0; i < 4; ++i) td[i] = readTB(d_in, t);
  Lin hd0 = readLin(d_in, t, true);
  Lin hd1 = readLin(d_in, t, true);
  Lin hd2 = readLin(d_in, t, true);
  float* OUT = (float*)d_out;

  // ---- carve workspace (all L2-resident, ~45 MB) ----
  size_t off = 0;
  auto carve = [&](size_t bytes) -> void* {
    void* p = (char*)d_ws + off;
    off += (bytes + 255) & ~(size_t)255;
    return p;
  };
  const size_t FB = 524288;               // max N*C floats (8192*64)
  const size_t EB = 2228224;              // max E*C / E*64 floats (34816*64)
  float* posA   = (float*)carve(8192 * 3 * sizeof(float));
  float* posBv  = (float*)carve(8192 * 3 * sizeof(float));
  float* bufX   = (float*)carve(FB * sizeof(float));
  Bufs B;
  B.h      = (float*)carve(FB * sizeof(float));
  B.adst   = (float*)carve(FB * sizeof(float));
  B.asrc   = (float*)carve(FB * sizeof(float));
  B.vv     = (float*)carve(FB * sizeof(float));
  B.conv   = (float*)carve(FB * sizeof(float));
  B.eHid   = (float*)carve(EB * sizeof(float));
  B.eDelta = (float*)carve(EB * sizeof(float));
  B.eAttn  = (float*)carve(EB * sizeof(float));
  B.Abf    = (__bf16*)carve(EB * sizeof(__bf16));       // bf16 A scratch
  B.BTbf   = (__bf16*)carve(512 * 512 * sizeof(__bf16)); // bf16 B^T scratch
  int* nbr17  = (int*)carve(8192 * 17 * sizeof(int));
  int* tdnbr  = (int*)carve(2048 * 16 * sizeof(int));
  int* sel    = (int*)carve(2048 * sizeof(int));
  float* bnMean = (float*)carve(512 * sizeof(float));
  float* bnInv  = (float*)carve(512 * sizeof(float));

  const int dims[5] = {32, 64, 128, 256, 512};
  int N = 8192;
  const float* curPos = POS;

  // ---- mlp_input: Lin(3->32) + BN(train stats) + ReLU ----
  lin3_kernel<<<gdiv(N * 32), 256, 0, stream>>>(X0, mi.lin.W, mi.lin.b, B.h, N, 32);
  bn_stats_kernel<<<32, 256, 0, stream>>>(B.h, N, 32, bnMean, bnInv);
  bn_relu_kernel<<<gdiv(N * 32), 256, 0, stream>>>(B.h, bnMean, bnInv,
                                                   mi.gamma, mi.beta, bufX, N, 32);

  // ---- t_in ----
  knn_kernel<<<gdiv(N), 256, 0, stream>>>(N, curPos, N, curPos, nbr17, 17, 1, 1);
  run_tb(tin, bufX, curPos, nbr17, N, 32, B, stream);

  // ---- 4 x (transition_down + transformer) ----
  for (int i = 0; i < 4; ++i) {
    const int C = dims[i], C2 = dims[i + 1], M = N / 4;
    // mlp on full cloud
    gemm(bufX, md[i].lin.W, md[i].lin.b, nullptr, B.h, N, C, C2, GF_BIAS, true, B, stream);
    bn_stats_kernel<<<C2, 256, 0, stream>>>(B.h, N, C2, bnMean, bnInv);
    bn_relu_kernel<<<gdiv(N * C2), 256, 0, stream>>>(B.h, bnMean, bnInv,
                                                     md[i].gamma, md[i].beta,
                                                     B.h, N, C2);
    // fps + gather
    fps_kernel<<<1, 1024, 0, stream>>>(curPos, N, M, sel);
    float* newPos = (i & 1) ? posBv : posA;
    gather_pos_kernel<<<gdiv(M), 256, 0, stream>>>(curPos, sel, M, newPos);
    // knn(full cloud, query = sampled) and per-row max pooling
    knn_kernel<<<gdiv(M), 256, 0, stream>>>(M, newPos, N, curPos, tdnbr, 16, 0, 0);
    gather_max_kernel<<<gdiv(M * C2), 256, 0, stream>>>(B.h, tdnbr, M, C2, bufX);
    // new self-knn graph on sampled cloud
    knn_kernel<<<gdiv(M), 256, 0, stream>>>(M, newPos, M, newPos, nbr17, 17, 1, 1);
    curPos = newPos;
    N = M;
    run_tb(td[i], bufX, curPos, nbr17, N, C2, B, stream);
  }

  // ---- global mean pool + head + log_softmax ----
  head_kernel<<<1, 512, 0, stream>>>(bufX, N, hd0.W, hd0.b, hd1.W, hd1.b,
                                     hd2.W, hd2.b, OUT);
}
